// GraphODEMD_2619930050781
// MI455X (gfx1250) — matrix operand order repaired and checked by reading, not compiled, not run on hardware
//
#include <hip/hip_runtime.h>

typedef _Float16 half8  __attribute__((ext_vector_type(8)));
typedef _Float16 half16 __attribute__((ext_vector_type(16)));
typedef float    float8 __attribute__((ext_vector_type(8)));

#define N_NODES 50000
#define N_EDGES 800000
#define TSTEPS  8

__device__ __forceinline__ float silu_f(float v) { return v / (1.0f + __expf(-v)); }

__device__ __forceinline__ half16 cat16(half8 lo, half8 hi) {
    return __builtin_shufflevector(lo, hi, 0,1,2,3,4,5,6,7,8,9,10,11,12,13,14,15);
}

__device__ __forceinline__ float8 wmma16(half16 a, half16 b, float8 c) {
    // D = A(16x32 f16) * B(32x16 f16) + C(16x16 f32)
    return __builtin_amdgcn_wmma_f32_16x16x32_f16(false, a, false, b, (short)0, c, false, false);
}

// ---------------------------------------------------------------------------
// Edge message MLP (two layers) with WMMA, fused SiLU + atomic segment-sum.
// ENC=true adds the radial feature and the coordinate update.
// Tile = 16 edges per wave. A: [16 x Kpad(160)] f16 in LDS.
// Weights staged transposed: Wt[n][k], stride AS/MS halves (16B aligned rows).
// ---------------------------------------------------------------------------
template<int KACT, bool ENC>
__global__ void __launch_bounds__(128) edge_mlp_kernel(
    const float* __restrict__ xpos, const float* __restrict__ hfeat,
    const int*   __restrict__ erow, const int*   __restrict__ ecol,
    const float* __restrict__ efea,
    const float* __restrict__ W1, const float* __restrict__ b1,
    const float* __restrict__ W2, const float* __restrict__ b2,
    const float* __restrict__ cW, const float* __restrict__ cb,
    float* __restrict__ agg, float* __restrict__ xagg, int nTiles)
{
    constexpr int AS = 168;  // K stride (halves): 5 chunks of 32, padded, bank-friendly
    constexpr int MS = 72;   // layer-2 K stride (halves): 2 chunks of 32

    __shared__ __attribute__((aligned(16))) _Float16 sW1[64 * AS];
    __shared__ __attribute__((aligned(16))) _Float16 sW2[64 * MS];
    __shared__ __attribute__((aligned(16))) _Float16 sA [4][16 * AS];
    __shared__ __attribute__((aligned(16))) _Float16 sM1[4][16 * MS];
    __shared__ float sB1[64], sB2[64], sCW[64];
    __shared__ float sCB;
    __shared__ int   sRow[4][16], sCol[4][16];
    __shared__ float sS[4][16];

    const int tid = threadIdx.x;
    // Stage transposed f16 weights (zero-padded K)
    for (int t = tid; t < 64 * AS; t += 128) {
        int k = t % AS, n = t / AS;
        sW1[t] = (k < KACT) ? (_Float16)W1[k * 64 + n] : (_Float16)0.0f;
    }
    for (int t = tid; t < 64 * MS; t += 128) {
        int k = t % MS, n = t / MS;
        sW2[t] = (k < 64) ? (_Float16)W2[k * 64 + n] : (_Float16)0.0f;
    }
    if (tid < 64) { sB1[tid] = b1[tid]; sB2[tid] = b2[tid]; if (ENC) sCW[tid] = cW[tid]; }
    if (ENC && tid == 0) sCB = cb[0];
    __syncthreads();

    const int wv = tid >> 5, lane = tid & 31;
    const int l16 = lane & 15, hihalf = lane >> 4;
    const int kbA = hihalf * 8;    // A-fragment K sub-offset per lane half
    const int kbB = hihalf * 16;   // unused; B fragments also use kbA
    _Float16* Aw = &sA[wv][0];
    _Float16* Mw = &sM1[wv][0];
    int* rowW = &sRow[wv][0];
    int* colW = &sCol[wv][0];

    for (int tile = blockIdx.x * 4 + wv; tile < nTiles; tile += gridDim.x * 4) {
        const int e0 = tile * 16;
        if (lane < 16) { rowW[lane] = erow[e0 + lane]; colW[lane] = ecol[e0 + lane]; }
        __builtin_amdgcn_wave_barrier();

        // Stage A = [h[row] | h[col] | (radial) | edge_fea | 0-pad] as f16
        for (int t = lane; t < 1024; t += 32) {
            int el = t >> 6, f = t & 63;
            Aw[el * AS + f] = (_Float16)hfeat[rowW[el] * 64 + f];
        }
        for (int t = lane; t < 1024; t += 32) {
            int el = t >> 6, f = t & 63;
            Aw[el * AS + 64 + f] = (_Float16)hfeat[colW[el] * 64 + f];
        }
        for (int t = lane; t < 512; t += 32) {
            int el = t >> 5, k = 128 + (t & 31);
            float v = 0.0f;
            if (ENC) {
                if (k == 128) {
                    int r = rowW[el], c = colW[el];
                    float dx = xpos[r*3+0] - xpos[c*3+0];
                    float dy = xpos[r*3+1] - xpos[c*3+1];
                    float dz = xpos[r*3+2] - xpos[c*3+2];
                    v = dx*dx + dy*dy + dz*dz;
                } else if (k < KACT) {
                    v = efea[(e0 + el) * 4 + (k - 129)];
                }
            } else {
                if (k < KACT) v = efea[(e0 + el) * 4 + (k - 128)];
            }
            Aw[el * AS + k] = (_Float16)v;
        }
        __builtin_amdgcn_wave_barrier();

        // ---- layer 1: [16 x 160] x [160 x 64], K-chunks of 32 ----
        float8 acc[4];
        #pragma unroll
        for (int nt = 0; nt < 4; nt++) {
            float bv = sB1[nt * 16 + l16];
            acc[nt] = (float8){bv, bv, bv, bv, bv, bv, bv, bv};
        }
        #pragma unroll
        for (int kc = 0; kc < 5; kc++) {
            const _Float16* ap = &Aw[l16 * AS + kc * 32 + kbA];
            half16 a = cat16(*(const half8*)ap, *(const half8*)(ap + 16));
            #pragma unroll
            for (int nt = 0; nt < 4; nt++) {
                const _Float16* bp = &sW1[(nt * 16 + l16) * AS + kc * 32 + kbA];
                half16 b = cat16(*(const half8*)bp, *(const half8*)(bp + 16));
                acc[nt] = wmma16(a, b, acc[nt]);
            }
        }
        __builtin_amdgcn_wave_barrier();
        // SiLU -> f16 A-layout in LDS (row = r + 8*hihalf == r + kbA)
        #pragma unroll
        for (int nt = 0; nt < 4; nt++)
            #pragma unroll
            for (int r = 0; r < 8; r++)
                Mw[(r + kbA) * MS + nt * 16 + l16] = (_Float16)silu_f(acc[nt][r]);
        __builtin_amdgcn_wave_barrier();

        // ---- layer 2: [16 x 64] x [64 x 64] ----
        float8 acc2[4];
        #pragma unroll
        for (int nt = 0; nt < 4; nt++) {
            float bv = sB2[nt * 16 + l16];
            acc2[nt] = (float8){bv, bv, bv, bv, bv, bv, bv, bv};
        }
        #pragma unroll
        for (int kc = 0; kc < 2; kc++) {
            const _Float16* ap = &Mw[l16 * MS + kc * 32 + kbA];
            half16 a = cat16(*(const half8*)ap, *(const half8*)(ap + 16));
            #pragma unroll
            for (int nt = 0; nt < 4; nt++) {
                const _Float16* bp = &sW2[(nt * 16 + l16) * MS + kc * 32 + kbA];
                half16 b = cat16(*(const half8*)bp, *(const half8*)(bp + 16));
                acc2[nt] = wmma16(a, b, acc2[nt]);
            }
        }
        // SiLU + atomic segment-sum of messages into agg[row]
        #pragma unroll
        for (int nt = 0; nt < 4; nt++)
            #pragma unroll
            for (int r = 0; r < 8; r++) {
                float v = silu_f(acc2[nt][r]);
                acc2[nt][r] = v;
                atomicAdd(&agg[rowW[r + kbA] * 64 + nt * 16 + l16], v);
            }

        if (ENC) {
            // s_e = m_e . cW + cb, butterfly-reduce over 16 feature lanes
            #pragma unroll
            for (int r = 0; r < 8; r++) {
                float p = 0.0f;
                #pragma unroll
                for (int nt = 0; nt < 4; nt++) p += acc2[nt][r] * sCW[nt * 16 + l16];
                #pragma unroll
                for (int m = 1; m < 16; m <<= 1) p += __shfl_xor(p, m, 32);
                if (l16 == 0) sS[wv][r + kbA] = p;
            }
            __builtin_amdgcn_wave_barrier();
            if (lane < 16) {
                float s = sS[wv][lane] + sCB;
                int r = rowW[lane], c = colW[lane];
                float dx = xpos[r*3+0] - xpos[c*3+0];
                float dy = xpos[r*3+1] - xpos[c*3+1];
                float dz = xpos[r*3+2] - xpos[c*3+2];
                atomicAdd(&xagg[r*3+0], dx * s);
                atomicAdd(&xagg[r*3+1], dy * s);
                atomicAdd(&xagg[r*3+2], dz * s);
            }
        }
        __builtin_amdgcn_wave_barrier();
    }
}

// ---------------------------------------------------------------------------
// Node MLP: u = silu([h|agg] @ W1 + b1); out = (base?) + u @ W2 + b2
// ---------------------------------------------------------------------------
__global__ void __launch_bounds__(128) node_mlp_kernel(
    const float* __restrict__ hin, const float* __restrict__ agg,
    const float* __restrict__ W1, const float* __restrict__ b1,
    const float* __restrict__ W2, const float* __restrict__ b2,
    const float* __restrict__ base, float* __restrict__ hout, int nTiles)
{
    constexpr int AS = 136;  // 4 K-chunks of 32
    constexpr int MS = 72;

    __shared__ __attribute__((aligned(16))) _Float16 sW1[64 * AS];
    __shared__ __attribute__((aligned(16))) _Float16 sW2[64 * MS];
    __shared__ __attribute__((aligned(16))) _Float16 sA [4][16 * AS];
    __shared__ __attribute__((aligned(16))) _Float16 sM1[4][16 * MS];
    __shared__ float sB1[64], sB2[64];

    const int tid = threadIdx.x;
    for (int t = tid; t < 64 * AS; t += 128) {
        int k = t % AS, n = t / AS;
        sW1[t] = (k < 128) ? (_Float16)W1[k * 64 + n] : (_Float16)0.0f;
    }
    for (int t = tid; t < 64 * MS; t += 128) {
        int k = t % MS, n = t / MS;
        sW2[t] = (k < 64) ? (_Float16)W2[k * 64 + n] : (_Float16)0.0f;
    }
    if (tid < 64) { sB1[tid] = b1[tid]; sB2[tid] = b2[tid]; }
    __syncthreads();

    const int wv = tid >> 5, lane = tid & 31;
    const int l16 = lane & 15, hihalf = lane >> 4;
    const int kbA = hihalf * 8, kbB = hihalf * 16;
    _Float16* Aw = &sA[wv][0];
    _Float16* Mw = &sM1[wv][0];

    for (int tile = blockIdx.x * 4 + wv; tile < nTiles; tile += gridDim.x * 4) {
        const int n0 = tile * 16;
        for (int t = lane; t < 1024; t += 32) {
            int el = t >> 6, f = t & 63;
            Aw[el * AS + f] = (_Float16)hin[(n0 + el) * 64 + f];
        }
        for (int t = lane; t < 1024; t += 32) {
            int el = t >> 6, f = t & 63;
            Aw[el * AS + 64 + f] = (_Float16)agg[(n0 + el) * 64 + f];
        }
        __builtin_amdgcn_wave_barrier();

        float8 acc[4];
        #pragma unroll
        for (int nt = 0; nt < 4; nt++) {
            float bv = sB1[nt * 16 + l16];
            acc[nt] = (float8){bv, bv, bv, bv, bv, bv, bv, bv};
        }
        #pragma unroll
        for (int kc = 0; kc < 4; kc++) {
            const _Float16* ap = &Aw[l16 * AS + kc * 32 + kbA];
            half16 a = cat16(*(const half8*)ap, *(const half8*)(ap + 16));
            #pragma unroll
            for (int nt = 0; nt < 4; nt++) {
                const _Float16* bp = &sW1[(nt * 16 + l16) * AS + kc * 32 + kbA];
                half16 b = cat16(*(const half8*)bp, *(const half8*)(bp + 16));
                acc[nt] = wmma16(a, b, acc[nt]);
            }
        }
        __builtin_amdgcn_wave_barrier();
        #pragma unroll
        for (int nt = 0; nt < 4; nt++)
            #pragma unroll
            for (int r = 0; r < 8; r++)
                Mw[(r + kbA) * MS + nt * 16 + l16] = (_Float16)silu_f(acc[nt][r]);
        __builtin_amdgcn_wave_barrier();

        float8 acc2[4];
        #pragma unroll
        for (int nt = 0; nt < 4; nt++) {
            float bv = sB2[nt * 16 + l16];
            acc2[nt] = (float8){bv, bv, bv, bv, bv, bv, bv, bv};
        }
        #pragma unroll
        for (int kc = 0; kc < 2; kc++) {
            const _Float16* ap = &Mw[l16 * MS + kc * 32 + kbA];
            half16 a = cat16(*(const half8*)ap, *(const half8*)(ap + 16));
            #pragma unroll
            for (int nt = 0; nt < 4; nt++) {
                const _Float16* bp = &sW2[(nt * 16 + l16) * MS + kc * 32 + kbA];
                half16 b = cat16(*(const half8*)bp, *(const half8*)(bp + 16));
                acc2[nt] = wmma16(a, b, acc2[nt]);
            }
        }
        #pragma unroll
        for (int nt = 0; nt < 4; nt++)
            #pragma unroll
            for (int r = 0; r < 8; r++) {
                int node = n0 + r + kbA;
                float v = acc2[nt][r];
                if (base) v += base[node * 64 + nt * 16 + l16];
                hout[node * 64 + nt * 16 + l16] = v;
            }
        __builtin_amdgcn_wave_barrier();
    }
}

// ---------------------------------------------------------------------------
// Small helper kernels
// ---------------------------------------------------------------------------
__global__ void embed_kernel(const float* __restrict__ hin, const float* __restrict__ W,
                             const float* __restrict__ b, float* __restrict__ hout) {
    int i = blockIdx.x * blockDim.x + threadIdx.x;
    if (i >= N_NODES * 64) return;
    int node = i >> 6, f = i & 63;
    float acc = b[f];
    #pragma unroll
    for (int k = 0; k < 16; k++) acc += hin[node * 16 + k] * W[k * 64 + f];
    hout[i] = acc;
}

__global__ void axpy_kernel(float* __restrict__ z, const float* __restrict__ x,
                            const float* __restrict__ y, float s, int n) {
    int i = blockIdx.x * blockDim.x + threadIdx.x;
    if (i < n) z[i] = x[i] + s * y[i];
}

__global__ void zero_kernel(float* __restrict__ p, int n) {
    int i = blockIdx.x * blockDim.x + threadIdx.x;
    if (i < n) p[i] = 0.0f;
}

__global__ void copy_kernel(float* __restrict__ dst, const float* __restrict__ src, int n) {
    int i = blockIdx.x * blockDim.x + threadIdx.x;
    if (i < n) dst[i] = src[i];
}

__global__ void __launch_bounds__(128) decode_kernel(
    const float* __restrict__ hin, const float* __restrict__ W1, const float* __restrict__ b1,
    const float* __restrict__ W2, const float* __restrict__ b2,
    const float* __restrict__ locm, float* __restrict__ xout) {
    int node = blockIdx.x * blockDim.x + threadIdx.x;
    if (node >= N_NODES) return;
    float hr[64];
    #pragma unroll
    for (int f = 0; f < 64; f++) hr[f] = hin[node * 64 + f];
    float a0 = 0.f, a1 = 0.f, a2 = 0.f;
    for (int j = 0; j < 64; j++) {
        float u = b1[j];
        #pragma unroll
        for (int f = 0; f < 64; f++) u += hr[f] * W1[f * 64 + j];
        u = silu_f(u);
        a0 += u * W2[j * 3 + 0];
        a1 += u * W2[j * 3 + 1];
        a2 += u * W2[j * 3 + 2];
    }
    xout[node * 3 + 0] = a0 + b2[0] + locm[node * 3 + 0];
    xout[node * 3 + 1] = a1 + b2[1] + locm[node * 3 + 1];
    xout[node * 3 + 2] = a2 + b2[2] + locm[node * 3 + 2];
}

// ---------------------------------------------------------------------------
extern "C" void kernel_launch(void* const* d_in, const int* in_sizes, int n_in,
                              void* d_out, int out_size, void* d_ws, size_t ws_size,
                              hipStream_t stream) {
    (void)in_sizes; (void)n_in; (void)out_size; (void)ws_size;
    const float* x_in = (const float*)d_in[0];
    const float* h_in = (const float*)d_in[1];
    const int*   ei   = (const int*)d_in[2];
    const float* ef   = (const float*)d_in[3];
    const float* locm = (const float*)d_in[4];
    const float* embW = (const float*)d_in[5];   const float* embB = (const float*)d_in[6];
    const float* eW1  = (const float*)d_in[7];   const float* eb1  = (const float*)d_in[8];
    const float* eW2  = (const float*)d_in[9];   const float* eb2  = (const float*)d_in[10];
    const float* cWt  = (const float*)d_in[11];  const float* cbt  = (const float*)d_in[12];
    const float* nW1  = (const float*)d_in[13];  const float* nb1  = (const float*)d_in[14];
    const float* nW2  = (const float*)d_in[15];  const float* nb2  = (const float*)d_in[16];
    const float* oeW1 = (const float*)d_in[17];  const float* oeb1 = (const float*)d_in[18];
    const float* oeW2 = (const float*)d_in[19];  const float* oeb2 = (const float*)d_in[20];
    const float* onW1 = (const float*)d_in[21];  const float* onb1 = (const float*)d_in[22];
    const float* onW2 = (const float*)d_in[23];  const float* onb2 = (const float*)d_in[24];
    const float* dW1  = (const float*)d_in[25];  const float* db1  = (const float*)d_in[26];
    const float* dW2  = (const float*)d_in[27];  const float* db2  = (const float*)d_in[28];

    const int* erow = ei;
    const int* ecol = ei + N_EDGES;
    float* out = (float*)d_out;

    const size_t NH = (size_t)N_NODES * 64;
    float* ws   = (float*)d_ws;
    float* hA   = ws;
    float* hB   = ws + NH;
    float* hC   = ws + 2 * NH;
    float* kb   = ws + 3 * NH;
    float* agg  = ws + 4 * NH;
    float* xcur = ws + 5 * NH;
    float* xagg = xcur + (size_t)N_NODES * 3;

    const int edgeTiles = N_EDGES / 16;   // 50000
    const int nodeTiles = N_NODES / 16;   // 3125
    const int edgeGrid = 3125, nodeGrid = 782;

    auto zero = [&](float* p, size_t n) {
        zero_kernel<<<(int)((n + 255) / 256), 256, 0, stream>>>(p, (int)n);
    };
    auto axpy = [&](float* z, const float* xs, const float* ys, float s) {
        axpy_kernel<<<(int)((NH + 255) / 256), 256, 0, stream>>>(z, xs, ys, s, (int)NH);
    };

    float* hcur = hA; float* htmp = hB; float* hacc = hC;

    // ---- embedding + encoder ----
    embed_kernel<<<(N_NODES * 64 + 255) / 256, 256, 0, stream>>>(h_in, embW, embB, hcur);
    copy_kernel<<<(N_NODES * 3 + 255) / 256, 256, 0, stream>>>(xcur, x_in, N_NODES * 3);

    for (int l = 0; l < 2; l++) {
        zero(agg, NH);
        zero(xagg, (size_t)N_NODES * 3);
        edge_mlp_kernel<133, true><<<edgeGrid, 128, 0, stream>>>(
            xcur, hcur, erow, ecol, ef,
            eW1 + (size_t)l * 133 * 64, eb1 + l * 64,
            eW2 + (size_t)l * 64 * 64,  eb2 + l * 64,
            cWt + l * 64, cbt + l, agg, xagg, edgeTiles);
        node_mlp_kernel<<<nodeGrid, 128, 0, stream>>>(
            hcur, agg,
            nW1 + (size_t)l * 128 * 64, nb1 + l * 64,
            nW2 + (size_t)l * 64 * 64,  nb2 + l * 64,
            hcur, htmp, nodeTiles);
        { float* t = hcur; hcur = htmp; htmp = t; }
        axpy_kernel<<<(N_NODES * 3 + 255) / 256, 256, 0, stream>>>(
            xcur, xcur, xagg, 1.0f, N_NODES * 3);
    }

    // decode t = 0
    decode_kernel<<<(N_NODES + 127) / 128, 128, 0, stream>>>(
        hcur, dW1, db1, dW2, db2, locm, out);

    // ---- RK4 ODE on h ----
    const float dt = 1.0f / 7.0f;
    auto rhs = [&](const float* hh) {
        zero(agg, NH);
        edge_mlp_kernel<132, false><<<edgeGrid, 128, 0, stream>>>(
            nullptr, hh, erow, ecol, ef,
            oeW1, oeb1, oeW2, oeb2, nullptr, nullptr, agg, nullptr, edgeTiles);
        node_mlp_kernel<<<nodeGrid, 128, 0, stream>>>(
            hh, agg, onW1, onb1, onW2, onb2, nullptr, kb, nodeTiles);
    };

    for (int st = 0; st < TSTEPS - 1; st++) {
        rhs(hcur);                                    // k1
        axpy(hacc, hcur, kb, dt / 6.0f);
        axpy(htmp, hcur, kb, dt * 0.5f);
        rhs(htmp);                                    // k2
        axpy(hacc, hacc, kb, dt / 3.0f);
        axpy(htmp, hcur, kb, dt * 0.5f);
        rhs(htmp);                                    // k3
        axpy(hacc, hacc, kb, dt / 3.0f);
        axpy(htmp, hcur, kb, dt);
        rhs(htmp);                                    // k4
        axpy(hacc, hacc, kb, dt / 6.0f);
        { float* t = hcur; hcur = hacc; hacc = t; }
        decode_kernel<<<(N_NODES + 127) / 128, 128, 0, stream>>>(
            hcur, dW1, db1, dW2, db2, locm, out + (size_t)(st + 1) * N_NODES * 3);
    }

    // outputs: dummy_v zeros, h_last, kld
    zero(out + (size_t)TSTEPS * N_NODES * 3, (size_t)TSTEPS * N_NODES * 3);
    copy_kernel<<<(int)((NH + 255) / 256), 256, 0, stream>>>(
        out + (size_t)2 * TSTEPS * N_NODES * 3, hcur, (int)NH);
    zero(out + (size_t)2 * TSTEPS * N_NODES * 3 + NH, 1);
}
